// model_49022756716820
// MI455X (gfx1250) — matrix operand order repaired and checked by reading, not compiled, not run on hardware
//
#include <hip/hip_runtime.h>
#include <math.h>

#define BB  8
#define NN  2048
#define KK  50
#define NBS 15
#define BNEPS 1e-5f

typedef _Float16 half_t;
typedef __attribute__((ext_vector_type(16))) _Float16 v16h;
typedef __attribute__((ext_vector_type(8)))  _Float16 v8h;
typedef __attribute__((ext_vector_type(8)))  float    v8f;
typedef __attribute__((ext_vector_type(4)))  float    v4f;

#define F_RELU 1
#define F_ACC  2
#define F_RAW  4
#define F_SIG  8

#define CDIV(a,b) (((a)+(b)-1)/(b))

// ---------------------------------------------------------------------------
// WMMA GEMM, fully padded & guard-free hot loop.
//   A (weights) : f16 [Mp, Kp] row-major, zero-padded.
//   B (acts)    : f16 [batch][Np, Kp]  (N-major, K-contiguous).
//   Y f16 out   : [batch][Np, Mp]  (same layout as next layer's B).
//   Y f32 out   : [batch][Np, Mp]  raw/accumulate path (conv11 segments).
// Per 32-K step: 2x b128 A-loads + 2x b128 B-loads + 1 WMMA.
// Epilogue: y = acc*esc[m] + esh[m] (bias+BN folded; pad rows have esc=esh=0),
// stored as one b128 f16 store per lane.
// ---------------------------------------------------------------------------
__global__ __launch_bounds__(256)
void wmma_gemm_kernel(const half_t* __restrict__ Ah,
                      const half_t* __restrict__ Bh,
                      half_t* __restrict__ Yh,
                      float* __restrict__ Yf,
                      const float* __restrict__ esc,
                      const float* __restrict__ esh,
                      int Mp, int Kp, int Np, int flags)
{
  const int wave = threadIdx.x >> 5;
  const int lane = threadIdx.x & 31;
  const int tiles_n = Np >> 4;
  const int tiles_m = Mp >> 4;
  const long tile = (long)blockIdx.x * 8 + wave;
  if (tile >= (long)tiles_m * tiles_n) return;       // wave-uniform
  const int tm = (int)(tile / tiles_n), tn = (int)(tile % tiles_n);
  const int mt = tm << 4, nt = tn << 4;
  const int r  = lane & 15, kh = lane >> 4;

  // A fragment: row mt+r, K-halves 8*kh and 16+8*kh (ISA 16-bit A layout)
  const half_t* Arow = Ah + (size_t)(mt + r) * Kp + (kh << 3);
  // B fragment: column nt+r, K-halves 8*kh and 16+8*kh (same order as A)
  const half_t* Brow = Bh + ((size_t)blockIdx.y * Np + (nt + r)) * Kp + (kh << 3);

  v8f acc = {};
#pragma unroll 2
  for (int k0 = 0; k0 < Kp; k0 += 32) {
    v8h a0 = *(const v8h*)(Arow + k0);
    v8h a1 = *(const v8h*)(Arow + k0 + 16);
    v8h b0 = *(const v8h*)(Brow + k0);
    v8h b1 = *(const v8h*)(Brow + k0 + 16);
    __builtin_prefetch((const void*)(Brow + k0 + 64), 0, 0);
    v16h a = __builtin_shufflevector(a0, a1, 0,1,2,3,4,5,6,7,8,9,10,11,12,13,14,15);
    v16h b = __builtin_shufflevector(b0, b1, 0,1,2,3,4,5,6,7,8,9,10,11,12,13,14,15);
    acc = __builtin_amdgcn_wmma_f32_16x16x32_f16(false, a, false, b,
                                                 (short)0, acc, false, false);
  }

  // C/D: lane holds N = nt+r, M = mt + 8*kh + rr (rr = 0..7, contiguous)
  const size_t obase = ((size_t)blockIdx.y * Np + (nt + r)) * Mp + mt + (kh << 3);
  if (flags & F_RAW) {
    v4f lo = {acc[0], acc[1], acc[2], acc[3]};
    v4f hi = {acc[4], acc[5], acc[6], acc[7]};
    if (flags & F_ACC) {
      lo += *(const v4f*)(Yf + obase);
      hi += *(const v4f*)(Yf + obase + 4);
    }
    *(v4f*)(Yf + obase)     = lo;
    *(v4f*)(Yf + obase + 4) = hi;
  } else {
    const int mb = mt + (kh << 3);
    v8h o;
#pragma unroll
    for (int rr = 0; rr < 8; ++rr) {
      float v = acc[rr] * esc[mb + rr] + esh[mb + rr];
      if (flags & F_RELU) v = fmaxf(v, 0.f);
      o[rr] = (half_t)v;
    }
    *(v8h*)(Yh + obase) = o;
  }
}

// pack f32 weight [M, wld] -> zero-padded f16 [Mp, Kp]
__global__ void packw_kernel(const float* __restrict__ W, int wld,
                             half_t* __restrict__ out, int M, int Kd, int Kp, int total)
{
  int idx = blockIdx.x * blockDim.x + threadIdx.x;
  if (idx >= total) return;
  int m = idx / Kp, k = idx % Kp;
  out[idx] = (m < M && k < Kd) ? (half_t)W[(size_t)m * wld + k] : (half_t)0.f;
}

// fold bias+BN into scale/shift, zero-padded to Mp
__global__ void fold_kernel(const float* bias, const float* g, const float* be,
                            const float* mm, const float* vv,
                            float* __restrict__ esc, float* __restrict__ esh, int M, int Mp)
{
  int m = blockIdx.x * blockDim.x + threadIdx.x;
  if (m >= Mp) return;
  if (m < M) {
    float sc = g[m] * rsqrtf(vv[m] + BNEPS);
    esc[m] = sc;
    esh[m] = bias[m] * sc + be[m] - mm[m] * sc;
  } else { esc[m] = 0.f; esh[m] = 0.f; }
}

// point cloud [B*N,3] f32 -> padded f16 [B*N, 32]
__global__ void pack_pc_kernel(const float* __restrict__ pc, half_t* __restrict__ out)
{
  int idx = blockIdx.x * blockDim.x + threadIdx.x;
  if (idx >= BB * NN * 32) return;
  int n = idx >> 5, k = idx & 31;
  out[idx] = (k < 3) ? (half_t)pc[n * 3 + k] : (half_t)0.f;
}

// Y[s,m] = max_{n<Nc} (float)X[s*sstride + n*rstride + m]
__global__ void colmax_f16_kernel(const half_t* __restrict__ X, float* __restrict__ Y,
                                  long sstride, int rstride, int M, int Nc, int S)
{
  int idx = blockIdx.x * blockDim.x + threadIdx.x;
  if (idx >= S * M) return;
  int s = idx / M, m = idx % M;
  const half_t* x = X + (size_t)s * (size_t)sstride + m;
  float mx = -3.4e38f;
  for (int n = 0; n < Nc; ++n) mx = fmaxf(mx, (float)x[(size_t)n * rstride]);
  Y[idx] = mx;
}

// FC (length-1 conv), f32 path
__global__ void fc_kernel(const float* __restrict__ W, int wld,
                          const float* __restrict__ X, long xstride,
                          float* __restrict__ Y, long ystride,
                          const float* bias, const float* bnG, const float* bnB,
                          const float* bnM, const float* bnV,
                          int M, int Kd, int S, int flags)
{
  int idx = blockIdx.x * blockDim.x + threadIdx.x;
  if (idx >= S * M) return;
  int s = idx / M, m = idx % M;
  const float* w = W + (size_t)m * (size_t)wld;
  const float* x = X + (size_t)s * (size_t)xstride;
  float acc = 0.f;
  for (int k = 0; k < Kd; ++k) acc += w[k] * x[k];
  float out;
  if (flags & F_RAW) out = acc;
  else {
    float bi = bias ? bias[m] : 0.f;
    if (bnG) {
      float sc = bnG[m] * rsqrtf(bnV[m] + BNEPS);
      out = (acc + bi) * sc + bnB[m] - bnM[m] * sc;
    } else out = acc + bi;
    if (flags & F_RELU) out = fmaxf(out, 0.f);
    if (flags & F_SIG)  out = 1.f / (1.f + expf(-out));
  }
  Y[(size_t)s * (size_t)ystride + m] = out;
}

// conv11 epilogue: (raw + gpart)*sc + sh, ReLU, emit packed f16 [B*N,128]
__global__ void conv11_epi_kernel(const float* __restrict__ raw, const float* __restrict__ gpart,
                                  const float* __restrict__ esc, const float* __restrict__ esh,
                                  half_t* __restrict__ outh)
{
  int idx = blockIdx.x * blockDim.x + threadIdx.x;
  if (idx >= BB * NN * 128) return;
  int b = idx / (NN * 128);
  int m = idx % 128;
  float v = (raw[idx] + gpart[b * 128 + m]) * esc[m] + esh[m];
  outh[idx] = (half_t)fmaxf(v, 0.f);
}

// in21 for one batch: [KK][NN][96] = concat(src_out(64), w_pc(1), key_pts(3), 0-pad)
__global__ void build_in21h_kernel(const half_t* __restrict__ so, const float* __restrict__ wpc,
                                   const float* __restrict__ kp, half_t* __restrict__ out)
{
  int idx = blockIdx.x * blockDim.x + threadIdx.x;
  if (idx >= KK * NN * 96) return;
  int k = idx / (NN * 96);
  int n = (idx / 96) % NN;
  int row = idx % 96;
  half_t v;
  if (row < 64)       v = so[(size_t)n * 64 + row];
  else if (row == 64) v = (half_t)wpc[(size_t)n * KK + k];
  else if (row < 68)  v = (half_t)kp[k * 3 + row - 65];
  else                v = (half_t)0.f;
  out[idx] = v;
}

__global__ void build_in31_kernel(const float* kf, const float* kp, float* out)
{
  int idx = blockIdx.x * blockDim.x + threadIdx.x;
  if (idx >= BB * KK * 67) return;
  int s = idx / 67, row = idx % 67;
  out[idx] = (row < 64) ? kf[s * 64 + row] : kp[s * 3 + row - 64];
}

// basis[b,r,k*3+j] = o33[(b*K+k)*45 + j*NBS + r], normalized over 150
__global__ void basis_kernel(const float* __restrict__ o33, float* __restrict__ basis)
{
  int idx = blockIdx.x * blockDim.x + threadIdx.x;
  if (idx >= BB * NBS) return;
  int b = idx / NBS, r = idx % NBS;
  float* dst = basis + (size_t)idx * 150;
  float ss = 0.f;
  for (int k = 0; k < KK; ++k)
    for (int j = 0; j < 3; ++j) {
      float v = o33[((size_t)(b * KK + k)) * 45 + j * NBS + r];
      dst[k * 3 + j] = v;
      ss += v * v;
    }
  float inv = 1.f / sqrtf(ss);
  for (int i = 0; i < 150; ++i) dst[i] *= inv;
}

// in71: [120][64][96]: rows = key_fea(64) | key_pts(3) | basis(3) | 0; cols>=KK -> 0
__global__ void build_in71h_kernel(const float* kf, const float* kp, const float* basis,
                                   half_t* __restrict__ out)
{
  int idx = blockIdx.x * blockDim.x + threadIdx.x;
  if (idx >= BB * NBS * 64 * 96) return;
  int s = idx / (64 * 96);
  int n = (idx / 96) % 64;          // keypoint column (padded to 64)
  int row = idx % 96;
  int b = s / NBS;
  float v = 0.f;
  if (n < KK) {
    if (row < 64)      v = kf[(b * KK + n) * 64 + row];
    else if (row < 67) v = kp[(b * KK + n) * 3 + row - 64];
    else if (row < 70) v = basis[(size_t)s * 150 + n * 3 + row - 67];
  }
  out[idx] = (half_t)v;
}

// conv73 (16->2, bias only) fused with max over K, *0.01, signs [-1,1]
__global__ void conv73max_kernel(const half_t* __restrict__ h72, const float* W,
                                 const float* bias, float* __restrict__ crange)
{
  int idx = blockIdx.x * blockDim.x + threadIdx.x;
  if (idx >= BB * NBS * 2) return;
  int s = idx / 2, o = idx % 2;
  float mx = -3.4e38f;
  for (int k = 0; k < KK; ++k) {
    float a = bias[o];
    const half_t* col = h72 + ((size_t)s * 64 + k) * 16;
    for (int c = 0; c < 16; ++c) a += W[o * 16 + c] * (float)col[c];
    mx = fmaxf(mx, a);
  }
  float cr = mx * 0.01f;
  crange[idx] = (o == 0) ? -cr : cr;
}

__global__ void build_in41_kernel(const float* kf, const float* gs, const float* gt,
                                  const float* kp, float* out)
{
  int idx = blockIdx.x * blockDim.x + threadIdx.x;
  if (idx >= BB * KK * 2115) return;
  int s = idx / 2115, row = idx % 2115;
  int b = s / KK;
  float v;
  if (row < 64)        v = kf[s * 64 + row];
  else if (row < 1088) v = gs[b * 1024 + row - 64];
  else if (row < 2112) v = gt[b * 1024 + row - 1088];
  else                 v = kp[s * 3 + row - 2112];
  out[idx] = v;
}

// in51: [120][64][160]: rows = kf2(128) | basis(3) | key_pts(3) | 0; cols>=KK -> 0
__global__ void build_in51h_kernel(const float* kf2, const float* basis, const float* kp,
                                   half_t* __restrict__ out)
{
  int idx = blockIdx.x * blockDim.x + threadIdx.x;
  if (idx >= BB * NBS * 64 * 160) return;
  int s = idx / (64 * 160);
  int n = (idx / 160) % 64;
  int row = idx % 160;
  int b = s / NBS;
  float v = 0.f;
  if (n < KK) {
    if (row < 128)      v = kf2[((size_t)(b * KK + n)) * 128 + row];
    else if (row < 131) v = basis[(size_t)s * 150 + n * 3 + row - 128];
    else if (row < 134) v = kp[(b * KK + n) * 3 + row - 131];
  }
  out[idx] = (half_t)v;
}

__global__ void build_in61_kernel(const float* net, const float* cr, float* out)
{
  int idx = blockIdx.x * blockDim.x + threadIdx.x;
  if (idx >= BB * NBS * 130) return;
  int s = idx / 130, row = idx % 130;
  out[idx] = (row < 128) ? net[s * 128 + row] : cr[s * 2 + row - 128];
}

__global__ void coef_kernel(const float* sig, const float* crange, const float* ratio,
                            float* coef_out, float* sample_coef)
{
  int idx = blockIdx.x * blockDim.x + threadIdx.x;
  if (idx >= BB * NBS) return;
  float c = sig[idx], rr = ratio[idx];
  float c0 = crange[idx * 2], c1 = crange[idx * 2 + 1];
  coef_out[idx]    = c * c0 + (1.f - c) * c1;
  sample_coef[idx] = rr * c0 + (1.f - rr) * c1;
}

__global__ void defkp_kernel(const float* kp, const float* basis, const float* coef,
                             const float* scoef, float* dkp, float* sdkp)
{
  int idx = blockIdx.x * blockDim.x + threadIdx.x;
  if (idx >= BB * KK * 3) return;
  int b = idx / (KK * 3);
  int rem = idx % (KK * 3);
  float a = 0.f, sa = 0.f;
  for (int r = 0; r < NBS; ++r) {
    float bv = basis[((size_t)(b * NBS + r)) * 150 + rem];
    a  += coef[b * NBS + r]  * bv;
    sa += scoef[b * NBS + r] * bv;
  }
  dkp[idx]  = kp[idx] + a;
  sdkp[idx] = kp[idx] + sa;
}

__global__ void defpc_kernel(const float* wpc, const float* dkp, const float* sdkp,
                             float* dpc, float* sdpc)
{
  int idx = blockIdx.x * blockDim.x + threadIdx.x;
  if (idx >= BB * NN * 3) return;
  int b = idx / (NN * 3);
  int n = (idx / 3) % NN;
  int j = idx % 3;
  const float* w = wpc + ((size_t)b * NN + n) * KK;
  float a = 0.f, sa = 0.f;
  for (int k = 0; k < KK; ++k) {
    float wv = w[k];
    a  += wv * dkp[(b * KK + k) * 3 + j];
    sa += wv * sdkp[(b * KK + k) * 3 + j];
  }
  dpc[idx] = a;
  sdpc[idx] = sa;
}

// per-point NN squared distance (optional x-mirror per side)
__global__ void chamfer_kernel(const float* __restrict__ A, const float* __restrict__ Bp,
                               float sxA, float sxB, float* __restrict__ minbuf)
{
  int idx = blockIdx.x * blockDim.x + threadIdx.x;
  if (idx >= BB * NN) return;
  int b = idx / NN, n = idx % NN;
  const float* pa = A + ((size_t)b * NN + n) * 3;
  float ax = pa[0] * sxA, ay = pa[1], az = pa[2];
  const float* pb = Bp + (size_t)b * NN * 3;
  float mn = 3.4e38f;
  for (int m = 0; m < NN; ++m) {
    float dx = ax - pb[0] * sxB, dy = ay - pb[1], dz = az - pb[2];
    mn = fminf(mn, dx * dx + dy * dy + dz * dz);
    pb += 3;
  }
  minbuf[idx] = mn;
}

__global__ void reduce_add_kernel(const float* __restrict__ X, int n,
                                  float* __restrict__ out, float scale)
{
  __shared__ float sm[256];
  float a = 0.f;
  for (int i = threadIdx.x; i < n; i += 256) a += X[i];
  sm[threadIdx.x] = a;
  __syncthreads();
  for (int s = 128; s > 0; s >>= 1) {
    if (threadIdx.x < s) sm[threadIdx.x] += sm[threadIdx.x + s];
    __syncthreads();
  }
  if (threadIdx.x == 0) out[0] += sm[0] * scale;   // single block: deterministic
}

__global__ void zero2_kernel(float* a, float* b)
{
  if (threadIdx.x == 0 && blockIdx.x == 0) { *a = 0.f; *b = 0.f; }
}

// ---------------------------------------------------------------------------
extern "C" void kernel_launch(void* const* d_in, const int* in_sizes, int n_in,
                              void* d_out_, int out_size, void* d_ws, size_t ws_size,
                              hipStream_t stream)
{
  (void)in_sizes; (void)n_in; (void)out_size; (void)ws_size;

  const float* src_pc  = (const float*)d_in[0];
  const float* tar_pc  = (const float*)d_in[1];
  const float* key_pts = (const float*)d_in[2];
  const float* w_pc    = (const float*)d_in[3];
  const float* ratio   = (const float*)d_in[4];

  const float *cW[24], *cB[24];
  for (int i = 0; i < 24; ++i) { cW[i] = (const float*)d_in[5 + 2*i]; cB[i] = (const float*)d_in[6 + 2*i]; }
  const float *bnG[21], *bnB[21], *bnM[21], *bnV[21];
  for (int i = 0; i < 21; ++i) {
    int b0 = 53 + 4*i;
    bnG[i] = (const float*)d_in[b0];   bnB[i] = (const float*)d_in[b0+1];
    bnM[i] = (const float*)d_in[b0+2]; bnV[i] = (const float*)d_in[b0+3];
  }
  enum {Cp1,Cp2,Cp3,Cp4,C11,C12,C13,C21,C22,C31,C32,C33,C41,C42,C43,C51,C52,C53,C61,C62,C63,C71,C72,C73};
  enum {Bp1,Bp2,Bp3,Bp4,B11,B12,B13,B21,B22,B31,B32,B41,B42,B43,B51,B52,B53,B61,B62,B71,B72};

  // ---- workspace carve (64B aligned)
  char* base = (char*)d_ws;
  size_t off = 0;
  auto a16 = [&](size_t n) { half_t* p = (half_t*)(base + off); off += ((n*2 + 63)/64)*64; return p; };
  auto a32 = [&](size_t n) { float*  p = (float*) (base + off); off += ((n*4 + 63)/64)*64; return p; };

  // f16 activation buffers ([Np, Kp] per sample)
  half_t* pch     = a16((size_t)BB*NN*32);
  half_t* f1h     = a16((size_t)BB*NN*64);
  half_t* f2h     = a16((size_t)BB*NN*256);
  half_t* f3h     = a16((size_t)BB*NN*512);
  half_t* f4h     = a16((size_t)BB*NN*1024);
  half_t* t128h   = a16((size_t)BB*NN*128);
  half_t* t64ah   = a16((size_t)BB*NN*64);
  half_t* srcouth = a16((size_t)BB*NN*64);
  half_t* in21h   = a16((size_t)KK*NN*96);
  half_t* z21h    = a16((size_t)KK*NN*64);
  half_t* z22h    = a16((size_t)KK*NN*64);
  half_t* in71h   = a16((size_t)BB*NBS*64*96);
  half_t* h71h    = a16((size_t)BB*NBS*64*32);
  half_t* h72h    = a16((size_t)BB*NBS*64*16);
  half_t* in51h   = a16((size_t)BB*NBS*64*160);
  half_t* h51h    = a16((size_t)BB*NBS*64*256);
  half_t* h52h    = a16((size_t)BB*NBS*64*128);
  half_t* h53h    = a16((size_t)BB*NBS*64*128);

  // f16 packed weights [Mp, Kp]
  half_t* wp1   = a16(64*32);     half_t* wp2   = a16(256*64);
  half_t* wp3   = a16(512*256);   half_t* wp4   = a16((size_t)1024*512);
  half_t* w11x  = a16(128*32);    half_t* w11f1 = a16(128*64);
  half_t* w11f2 = a16(128*256);   half_t* w11f3 = a16(128*512);
  half_t* w11f4 = a16(128*1024);
  half_t* w12   = a16(64*128);    half_t* w13   = a16(64*64);
  half_t* w21   = a16(64*96);     half_t* w22   = a16(64*64);
  half_t* w51   = a16(256*160);   half_t* w52   = a16(128*256);
  half_t* w53   = a16(128*128);   half_t* w71   = a16(32*96);
  half_t* w72   = a16(16*32);

  // folded epilogues (scale, shift) padded to Mp
  float* eP1 = a32(2*64);   float* eP2 = a32(2*256);
  float* eP3 = a32(2*512);  float* eP4 = a32(2*1024);
  float* e11 = a32(2*128);  float* e12 = a32(2*64);  float* e13 = a32(2*64);
  float* e21 = a32(2*64);   float* e22 = a32(2*64);
  float* e51 = a32(2*256);  float* e52 = a32(2*128); float* e53 = a32(2*128);
  float* e71 = a32(2*32);   float* e72 = a32(2*16);

  // f32 buffers
  float* t128f  = a32((size_t)BB*NN*128);
  float* gsrc   = a32(BB*1024);
  float* gtar   = a32(BB*1024);
  float* gpart  = a32(BB*128);
  float* keyfea = a32(BB*KK*64);
  float* in31   = a32(BB*KK*67);
  float* h31    = a32(BB*KK*256);
  float* h32    = a32(BB*KK*512);
  float* o33    = a32(BB*KK*45);
  float* in41   = a32((size_t)BB*KK*2115);
  float* h41    = a32(BB*KK*256);
  float* h42    = a32(BB*KK*128);
  float* kf2    = a32(BB*KK*128);
  float* net53  = a32(BB*NBS*128);
  float* in61   = a32(BB*NBS*130);
  float* h61    = a32(BB*NBS*64);
  float* h62    = a32(BB*NBS*32);
  float* sig    = a32(BB*NBS);
  float* scoef  = a32(BB*NBS);
  float* sdpc   = a32((size_t)BB*NN*3);
  float* minbuf = a32((size_t)BB*NN);

  // ---- output layout (flat concat, reference return order)
  float* out = (float*)d_out_;
  float* o_dkp   = out;            // [B,K,3]    1200
  float* o_dpc   = out + 1200;     // [B,N,3]    49152
  float* o_cd    = out + 50352;    // scalar
  float* o_basis = out + 50353;    // [B,15,150] 18000
  float* o_coef  = out + 68353;    // [B,1,15]   120
  float* o_sdkp  = out + 68473;    // [B,K,3]    1200
  float* o_sym   = out + 69673;    // scalar
  float* o_cr    = out + 69674;    // [B,15,2]   240

  auto packw = [&](const float* W, int wld, half_t* dst, int M, int Kd, int Kp, int Mp) {
    int total = Mp * Kp;
    packw_kernel<<<CDIV(total,256),256,0,stream>>>(W, wld, dst, M, Kd, Kp, total);
  };
  auto fold = [&](int c, int bn, float* e, int M, int Mp) {
    fold_kernel<<<CDIV(Mp,256),256,0,stream>>>(cB[c], bnG[bn], bnB[bn], bnM[bn], bnV[bn],
                                               e, e + Mp, M, Mp);
  };
  auto gemm = [&](const half_t* A, const half_t* B, half_t* Yh, float* Yf,
                  const float* e, int Mp, int Kp, int Np, int nb, int flags) {
    long tiles = (long)(Mp >> 4) * (Np >> 4);
    dim3 grid((unsigned)CDIV(tiles, 8), nb);
    wmma_gemm_kernel<<<grid,256,0,stream>>>(A, B, Yh, Yf, e, e ? e + Mp : nullptr,
                                            Mp, Kp, Np, flags);
  };
  auto fcl = [&](const float* W, int wld, const float* X, long xs, float* Y,
                 const float* bias, int bn, int M, int Kd, int S, int flags) {
    fc_kernel<<<CDIV(S*M,256),256,0,stream>>>(W, wld, X, xs, Y, (long)M, bias,
        bn >= 0 ? bnG[bn] : nullptr, bn >= 0 ? bnB[bn] : nullptr,
        bn >= 0 ? bnM[bn] : nullptr, bn >= 0 ? bnV[bn] : nullptr, M, Kd, S, flags);
  };

  // ---- pack weights + fold epilogues
  packw(cW[Cp1], 3,    wp1,   64,   3,    32,   64);
  packw(cW[Cp2], 64,   wp2,   256,  64,   64,   256);
  packw(cW[Cp3], 256,  wp3,   512,  256,  256,  512);
  packw(cW[Cp4], 512,  wp4,   1024, 512,  512,  1024);
  packw(cW[C11]+0,   2883, w11x,  128, 3,    32,   128);
  packw(cW[C11]+3,   2883, w11f1, 128, 64,   64,   128);
  packw(cW[C11]+67,  2883, w11f2, 128, 256,  256,  128);
  packw(cW[C11]+323, 2883, w11f3, 128, 512,  512,  128);
  packw(cW[C11]+835, 2883, w11f4, 128, 1024, 1024, 128);
  packw(cW[C12], 128,  w12, 64,  128, 128, 64);
  packw(cW[C13], 64,   w13, 64,  64,  64,  64);
  packw(cW[C21], 68,   w21, 64,  68,  96,  64);
  packw(cW[C22], 64,   w22, 64,  64,  64,  64);
  packw(cW[C51], 134,  w51, 256, 134, 160, 256);
  packw(cW[C52], 256,  w52, 128, 256, 256, 128);
  packw(cW[C53], 128,  w53, 128, 128, 128, 128);
  packw(cW[C71], 70,   w71, 32,  70,  96,  32);
  packw(cW[C72], 32,   w72, 16,  32,  32,  16);
  fold(Cp1,Bp1,eP1,64,64);   fold(Cp2,Bp2,eP2,256,256);
  fold(Cp3,Bp3,eP3,512,512); fold(Cp4,Bp4,eP4,1024,1024);
  fold(C11,B11,e11,128,128); fold(C12,B12,e12,64,64); fold(C13,B13,e13,64,64);
  fold(C21,B21,e21,64,64);   fold(C22,B22,e22,64,64);
  fold(C51,B51,e51,256,256); fold(C52,B52,e52,128,128); fold(C53,B53,e53,128,128);
  fold(C71,B71,e71,32,32);   fold(C72,B72,e72,16,16);

  // ---- PointNet(src)
  pack_pc_kernel<<<CDIV(BB*NN*32,256),256,0,stream>>>(src_pc, pch);
  gemm(wp1, pch, f1h, nullptr, eP1, 64,   32,  NN, BB, F_RELU);
  gemm(wp2, f1h, f2h, nullptr, eP2, 256,  64,  NN, BB, F_RELU);
  gemm(wp3, f2h, f3h, nullptr, eP3, 512,  256, NN, BB, F_RELU);
  gemm(wp4, f3h, f4h, nullptr, eP4, 1024, 512, NN, BB, F_RELU);
  colmax_f16_kernel<<<CDIV(BB*1024,256),256,0,stream>>>(f4h, gsrc, (long)NN*1024, 1024, 1024, NN, BB);

  // ---- conv11 as accumulated segment GEMMs (f32 raw) + global part + epilogue
  gemm(w11x,  pch, nullptr, t128f, nullptr, 128, 32,   NN, BB, F_RAW);
  gemm(w11f1, f1h, nullptr, t128f, nullptr, 128, 64,   NN, BB, F_RAW|F_ACC);
  gemm(w11f2, f2h, nullptr, t128f, nullptr, 128, 256,  NN, BB, F_RAW|F_ACC);
  gemm(w11f3, f3h, nullptr, t128f, nullptr, 128, 512,  NN, BB, F_RAW|F_ACC);
  gemm(w11f4, f4h, nullptr, t128f, nullptr, 128, 1024, NN, BB, F_RAW|F_ACC);
  fcl(cW[C11]+1859, 2883, gsrc, 1024, gpart, nullptr, -1, 128, 1024, BB, F_RAW);
  conv11_epi_kernel<<<CDIV(BB*NN*128,256),256,0,stream>>>(t128f, gpart, e11, e11+128, t128h);

  gemm(w12, t128h, t64ah,   nullptr, e12, 64, 128, NN, BB, F_RELU);
  gemm(w13, t64ah, srcouth, nullptr, e13, 64, 64,  NN, BB, F_RELU);

  // ---- PointNet(tar) -> gtar (reuse buffers)
  pack_pc_kernel<<<CDIV(BB*NN*32,256),256,0,stream>>>(tar_pc, pch);
  gemm(wp1, pch, f1h, nullptr, eP1, 64,   32,  NN, BB, F_RELU);
  gemm(wp2, f1h, f2h, nullptr, eP2, 256,  64,  NN, BB, F_RELU);
  gemm(wp3, f2h, f3h, nullptr, eP3, 512,  256, NN, BB, F_RELU);
  gemm(wp4, f3h, f4h, nullptr, eP4, 1024, 512, NN, BB, F_RELU);
  colmax_f16_kernel<<<CDIV(BB*1024,256),256,0,stream>>>(f4h, gtar, (long)NN*1024, 1024, 1024, NN, BB);

  // ---- stage 21/22 -> key_fea (chunked per batch)
  for (int b = 0; b < BB; ++b) {
    build_in21h_kernel<<<CDIV(KK*NN*96,256),256,0,stream>>>(
        srcouth + (size_t)b*NN*64, w_pc + (size_t)b*NN*KK, key_pts + (size_t)b*KK*3, in21h);
    gemm(w21, in21h, z21h, nullptr, e21, 64, 96, NN, KK, F_RELU);
    gemm(w22, z21h,  z22h, nullptr, e22, 64, 64, NN, KK, 0);
    colmax_f16_kernel<<<CDIV(KK*64,256),256,0,stream>>>(z22h, keyfea + (size_t)b*KK*64,
                                                        (long)NN*64, 64, 64, NN, KK);
  }

  // ---- stage 31-33 -> basis
  build_in31_kernel<<<CDIV(BB*KK*67,256),256,0,stream>>>(keyfea, key_pts, in31);
  fcl(cW[C31], 67,  in31, 67,  h31, cB[C31], B31, 256, 67,  BB*KK, F_RELU);
  fcl(cW[C32], 256, h31,  256, h32, cB[C32], B32, 512, 256, BB*KK, F_RELU);
  fcl(cW[C33], 512, h32,  512, o33, cB[C33], -1,  45,  512, BB*KK, 0);
  basis_kernel<<<CDIV(BB*NBS,64),64,0,stream>>>(o33, o_basis);

  // ---- stage 71-73 -> coef_range
  build_in71h_kernel<<<CDIV(BB*NBS*64*96,256),256,0,stream>>>(keyfea, key_pts, o_basis, in71h);
  gemm(w71, in71h, h71h, nullptr, e71, 32, 96, 64, BB*NBS, F_RELU);
  gemm(w72, h71h,  h72h, nullptr, e72, 16, 32, 64, BB*NBS, F_RELU);
  conv73max_kernel<<<CDIV(BB*NBS*2,64),64,0,stream>>>(h72h, cW[C73], cB[C73], o_cr);

  // ---- stage 41-43 -> kf2
  build_in41_kernel<<<CDIV(BB*KK*2115,256),256,0,stream>>>(keyfea, gsrc, gtar, key_pts, in41);
  fcl(cW[C41], 2115, in41, 2115, h41, cB[C41], B41, 256, 2115, BB*KK, F_RELU);
  fcl(cW[C42], 256,  h41,  256,  h42, cB[C42], B42, 128, 256,  BB*KK, F_RELU);
  fcl(cW[C43], 128,  h42,  128,  kf2, cB[C43], B43, 128, 128,  BB*KK, F_RELU);

  // ---- stage 51-53 -> net53
  build_in51h_kernel<<<CDIV(BB*NBS*64*160,256),256,0,stream>>>(kf2, o_basis, key_pts, in51h);
  gemm(w51, in51h, h51h, nullptr, e51, 256, 160, 64, BB*NBS, F_RELU);
  gemm(w52, h51h,  h52h, nullptr, e52, 128, 256, 64, BB*NBS, F_RELU);
  gemm(w53, h52h,  h53h, nullptr, e53, 128, 128, 64, BB*NBS, 0);
  colmax_f16_kernel<<<CDIV(BB*NBS*128,256),256,0,stream>>>(h53h, net53, 64*128, 128, 128, KK, BB*NBS);

  // ---- stage 61-63 -> coef
  build_in61_kernel<<<CDIV(BB*NBS*130,256),256,0,stream>>>(net53, o_cr, in61);
  fcl(cW[C61], 130, in61, 130, h61, cB[C61], B61, 64, 130, BB*NBS, F_RELU);
  fcl(cW[C62], 64,  h61,  64,  h62, cB[C62], B62, 32, 64,  BB*NBS, F_RELU);
  fcl(cW[C63], 32,  h62,  32,  sig, cB[C63], -1,  1,  32,  BB*NBS, F_SIG);
  coef_kernel<<<CDIV(BB*NBS,64),64,0,stream>>>(sig, o_cr, ratio, o_coef, scoef);

  // ---- deformation + losses
  defkp_kernel<<<CDIV(BB*KK*3,64),64,0,stream>>>(key_pts, o_basis, o_coef, scoef, o_dkp, o_sdkp);
  defpc_kernel<<<CDIV(BB*NN*3,256),256,0,stream>>>(w_pc, o_dkp, o_sdkp, o_dpc, sdpc);

  zero2_kernel<<<1,32,0,stream>>>(o_cd, o_sym);
  const float inv = 1.f / (float)(BB * NN);
  chamfer_kernel<<<CDIV(BB*NN,256),256,0,stream>>>(o_dpc, tar_pc, 1.f, 1.f, minbuf);
  reduce_add_kernel<<<1,256,0,stream>>>(minbuf, BB*NN, o_cd, inv);
  chamfer_kernel<<<CDIV(BB*NN,256),256,0,stream>>>(tar_pc, o_dpc, 1.f, 1.f, minbuf);
  reduce_add_kernel<<<1,256,0,stream>>>(minbuf, BB*NN, o_cd, inv);
  chamfer_kernel<<<CDIV(BB*NN,256),256,0,stream>>>(sdpc, sdpc, 1.f, -1.f, minbuf);
  reduce_add_kernel<<<1,256,0,stream>>>(minbuf, BB*NN, o_sym, inv);
  chamfer_kernel<<<CDIV(BB*NN,256),256,0,stream>>>(sdpc, sdpc, -1.f, 1.f, minbuf);
  reduce_add_kernel<<<1,256,0,stream>>>(minbuf, BB*NN, o_sym, inv);
}
